// MOARobertaLayer_67130338836512
// MI455X (gfx1250) — matrix operand order repaired and checked by reading, not compiled, not run on hardware
//
#include <hip/hip_runtime.h>
#include <hip/hip_bf16.h>

// ---------------------------------------------------------------------------
// MOA Roberta layer for MI455X (gfx1250, wave32, WMMA bf16 path)
//
// All GEMMs run as bf16 x bf16 -> f32 via v_wmma_f32_16x16x32_bf16.
// Fragment layouts follow CDNA5 ISA 7.12.2:
//   A (16x32 bf16): lane l -> row m=l&15, half h=l>>4;
//       VGPR i holds K pair: k = h*8 + (i&3)*2 + (i>=4 ? 16 : 0)
//   B (32x16 bf16): lane l -> col n=l&15, half h=l>>4;
//       VGPR i holds K pair: k = h*8 + (i&3)*2 + (i>=4 ? 16 : 0)   (we stage B transposed in LDS
//       so these pairs are contiguous 32-bit LDS reads)
//   C/D (16x16 f32): lane l -> col n=l&15; VGPR r -> row m = r + 8*(l>>4)
//
// Main GEMM: 128x64 block tile (8 waves, 32x32 per wave, 4 accumulators),
// double-buffered LDS staging so global loads for tile k+1 overlap the WMMA
// block on tile k (one barrier per K-iteration; LOADcnt drains under WMMA).
// ---------------------------------------------------------------------------

typedef __bf16 bf16;
typedef bf16  v16bf __attribute__((ext_vector_type(16)));
typedef bf16  v8bf  __attribute__((ext_vector_type(8)));
typedef float v8f   __attribute__((ext_vector_type(8)));

union FragU { v16bf v; unsigned int u[8]; };

__device__ __forceinline__ v16bf load_a_frag(const bf16* As, int stride,
                                             int m0, int kb, int lane) {
  FragU f;
  int m = lane & 15, h = lane >> 4;
  const bf16* base = As + (m0 + m) * stride + kb + h * 8;
#pragma unroll
  for (int i = 0; i < 8; ++i) {
    int k = ((i & 3) << 1) + ((i >> 2) << 4);
    f.u[i] = *(const unsigned int*)(base + k);
  }
  return f.v;
}

__device__ __forceinline__ v16bf load_b_frag(const bf16* Bt, int stride,
                                             int n0, int kb, int lane) {
  FragU f;
  int n = lane & 15, h = lane >> 4;
  const bf16* base = Bt + (n0 + n) * stride + kb + h * 8;
#pragma unroll
  for (int i = 0; i < 8; ++i) {
    f.u[i] = *(const unsigned int*)(base + ((i & 3) << 1) + ((i >> 2) << 4));
  }
  return f.v;
}

__device__ __forceinline__ v8f wmma_bf16(v16bf a, v16bf b, v8f c) {
  return __builtin_amdgcn_wmma_f32_16x16x32_bf16(
      /*neg_a=*/false, a, /*neg_b=*/false, b,
      /*c_mod=*/(short)0, c, /*reuse_a=*/false, /*reuse_b=*/false);
}

__device__ __forceinline__ float gelu_exact(float x) {
  return 0.5f * x * (1.0f + erff(x * 0.70710678118654752f));
}

// ---------------------------------------------------------------------------
// fp32 -> bf16 conversion
// ---------------------------------------------------------------------------
__global__ __launch_bounds__(256) void cvt_f32_bf16(const float* __restrict__ src,
                                                    bf16* __restrict__ dst, int n) {
  int i = blockIdx.x * blockDim.x + threadIdx.x;
  int stride = gridDim.x * blockDim.x;
  for (; i < n; i += stride) dst[i] = (bf16)src[i];
}

// ---------------------------------------------------------------------------
// Generic tiled GEMM: C[M,N] = act(A_bf16[M,K] @ B_bf16[K,N] + bias) (+ residual)
// Block tile 128x64, BK=64, double-buffered LDS. 8 waves; each wave computes
// a 32x32 sub-tile = 4 WMMA accumulators -> 8 WMMAs per wave per K-iteration.
// act: 0=none, 1=exact GELU. Writes f32 (Cf) and/or bf16 (Cb) outputs.
// ---------------------------------------------------------------------------
#define LDT 72  // LDS row stride (elements); multiple of 8 -> 16B-aligned rows

__global__ __launch_bounds__(256) void gemm_bf16(
    const bf16* __restrict__ A, const bf16* __restrict__ B,
    const float* __restrict__ bias, const float* __restrict__ residual,
    float* __restrict__ Cf, bf16* __restrict__ Cb,
    int M, int N, int K, int act) {
  __shared__ bf16 As[2][128 * LDT];
  __shared__ bf16 Bt[2][64 * LDT];

  int tid = threadIdx.x;
  int lane = tid & 31, w = tid >> 5;
  int m0 = (w >> 1) * 32, n0 = (w & 1) * 32;
  int mBase = blockIdx.y * 128, nBase = blockIdx.x * 64;

  v8f c00 = {}, c01 = {}, c10 = {}, c11 = {};

  // A staging: 128 rows x 64 cols; thread -> row tid>>1, 32 cols at (tid&1)*32
  int ra = tid >> 1;
  int ca = (tid & 1) * 32;
  // B staging: 64 K-rows x 64 cols; thread -> row tid>>2, 16 cols at (tid&3)*16
  int rb = tid >> 2;
  int cb = (tid & 3) * 16;

  const bf16* aBase = A + (size_t)(mBase + ra) * K + ca;
  const bf16* bBase = B + (size_t)rb * N + nBase + cb;

  // Prologue: stage tile 0 into buffer 0 (distinct regs -> loads pipeline)
  {
    v8bf a0 = *(const v8bf*)(aBase);
    v8bf a1 = *(const v8bf*)(aBase + 8);
    v8bf a2 = *(const v8bf*)(aBase + 16);
    v8bf a3 = *(const v8bf*)(aBase + 24);
    v8bf b0 = *(const v8bf*)(bBase);
    v8bf b1 = *(const v8bf*)(bBase + 8);
    bf16* Ad = &As[0][ra * LDT + ca];
    *(v8bf*)(Ad)      = a0;
    *(v8bf*)(Ad + 8)  = a1;
    *(v8bf*)(Ad + 16) = a2;
    *(v8bf*)(Ad + 24) = a3;
#pragma unroll
    for (int j = 0; j < 8; ++j) {
      Bt[0][(cb + j) * LDT + rb]     = b0[j];
      Bt[0][(cb + 8 + j) * LDT + rb] = b1[j];
    }
  }
  __syncthreads();

  int cur = 0;
  for (int kk = 0; kk < K; kk += 64) {
    bool hasNext = (kk + 64) < K;
    v8bf a0 = {}, a1 = {}, a2 = {}, a3 = {}, b0 = {}, b1 = {};
    if (hasNext) {
      // Issue next tile's global loads before the WMMA block so VMEM latency
      // overlaps matrix math (WMMA waits only on DScnt).
      const bf16* ag = aBase + kk + 64;
      a0 = *(const v8bf*)(ag);
      a1 = *(const v8bf*)(ag + 8);
      a2 = *(const v8bf*)(ag + 16);
      a3 = *(const v8bf*)(ag + 24);
      const bf16* bg = bBase + (size_t)(kk + 64) * N;
      b0 = *(const v8bf*)(bg);
      b1 = *(const v8bf*)(bg + 8);
      if (kk + 128 < K) {
        __builtin_prefetch(ag + 64, 0, 1);                 // tile after next (A)
        __builtin_prefetch(bg + (size_t)64 * N, 0, 1);     // tile after next (B)
      }
    }

    const bf16* Ac = As[cur];
    const bf16* Bc = Bt[cur];
#pragma unroll
    for (int ks = 0; ks < 2; ++ks) {
      v16bf fa0 = load_a_frag(Ac, LDT, m0,      ks * 32, lane);
      v16bf fa1 = load_a_frag(Ac, LDT, m0 + 16, ks * 32, lane);
      v16bf fb0 = load_b_frag(Bc, LDT, n0,      ks * 32, lane);
      v16bf fb1 = load_b_frag(Bc, LDT, n0 + 16, ks * 32, lane);
      c00 = wmma_bf16(fa0, fb0, c00);
      c01 = wmma_bf16(fa0, fb1, c01);
      c10 = wmma_bf16(fa1, fb0, c10);
      c11 = wmma_bf16(fa1, fb1, c11);
    }

    if (hasNext) {
      bf16* An = As[cur ^ 1];
      bf16* Bn = Bt[cur ^ 1];
      bf16* Ad = An + ra * LDT + ca;
      *(v8bf*)(Ad)      = a0;
      *(v8bf*)(Ad + 8)  = a1;
      *(v8bf*)(Ad + 16) = a2;
      *(v8bf*)(Ad + 24) = a3;
#pragma unroll
      for (int j = 0; j < 8; ++j) {
        Bn[(cb + j) * LDT + rb]     = b0[j];
        Bn[(cb + 8 + j) * LDT + rb] = b1[j];
      }
    }
    __syncthreads();
    cur ^= 1;
  }

  // Epilogue
  int n = lane & 15, h = lane >> 4;
#pragma unroll
  for (int rr = 0; rr < 8; ++rr) {
    int gm0 = mBase + m0 + rr + h * 8;   // rows of c00/c01
    int gm1 = gm0 + 16;                  // rows of c10/c11
    int gc0 = nBase + n0 + n;
    int gc1 = gc0 + 16;
    float v00 = c00[rr] + bias[gc0];
    float v01 = c01[rr] + bias[gc1];
    float v10 = c10[rr] + bias[gc0];
    float v11 = c11[rr] + bias[gc1];
    if (act == 1) {
      v00 = gelu_exact(v00); v01 = gelu_exact(v01);
      v10 = gelu_exact(v10); v11 = gelu_exact(v11);
    }
    size_t i00 = (size_t)gm0 * N + gc0;
    size_t i01 = (size_t)gm0 * N + gc1;
    size_t i10 = (size_t)gm1 * N + gc0;
    size_t i11 = (size_t)gm1 * N + gc1;
    if (residual) {
      v00 += residual[i00]; v01 += residual[i01];
      v10 += residual[i10]; v11 += residual[i11];
    }
    if (Cf) { Cf[i00] = v00; Cf[i01] = v01; Cf[i10] = v10; Cf[i11] = v11; }
    if (Cb) {
      Cb[i00] = (bf16)v00; Cb[i01] = (bf16)v01;
      Cb[i10] = (bf16)v10; Cb[i11] = (bf16)v11;
    }
  }
}

// ---------------------------------------------------------------------------
// Fused attention (flash-style): per (b, head, 64-query tile) block.
// S = (Q K^T) * 1/8 via WMMA; online softmax in LDS; O += P V via WMMA.
// Q/K/V/ctx all bf16 with layout [B*L, D], head slice at col head*64.
// ---------------------------------------------------------------------------
__global__ __launch_bounds__(256) void attn_kernel(
    const bf16* __restrict__ Q, const bf16* __restrict__ Kx,
    const bf16* __restrict__ V, bf16* __restrict__ ctx) {
  const int L = 1024, Dm = 1024, DH = 64;
  int qt = blockIdx.x;
  int bh = blockIdx.y;
  int b = bh >> 4, head = bh & 15;
  int qBase = qt * 64;

  __shared__ bf16 Qs[64 * LDT];
  __shared__ bf16 Ks[64 * LDT];
  __shared__ bf16 Vt[64 * LDT];
  __shared__ bf16 Ps[64 * 66];
  __shared__ float Ss[64 * 65];
  __shared__ float mrow[64], lrow[64], alphaS[64];
  __shared__ float redm[64 * 4], reds[64 * 4];

  int tid = threadIdx.x;
  int lane = tid & 31, w = tid >> 5;
  int m0 = (w >> 1) * 16, n0 = (w & 1) * 32;
  int r = tid >> 2, cq = (tid & 3) * 16;

  // Stage Q tile once
  {
    const bf16* qg = Q + (size_t)(b * L + qBase + r) * Dm + head * DH + cq;
    *(v8bf*)(Qs + r * LDT + cq)     = *(const v8bf*)qg;
    *(v8bf*)(Qs + r * LDT + cq + 8) = *(const v8bf*)(qg + 8);
  }
  if (tid < 64) { mrow[tid] = -3.0e38f; lrow[tid] = 0.f; }

  v8f o0 = {}, o1 = {};
  int n = lane & 15, h = lane >> 4;

  for (int kb = 0; kb < L; kb += 64) {
    __syncthreads();  // protect Ks/Vt/Ps reuse from previous iteration
    // Stage K tile (row-major == B-transposed for S = Q K^T)
    const bf16* kg = Kx + (size_t)(b * L + kb + r) * Dm + head * DH + cq;
    *(v8bf*)(Ks + r * LDT + cq)     = *(const v8bf*)kg;
    *(v8bf*)(Ks + r * LDT + cq + 8) = *(const v8bf*)(kg + 8);
    // Stage V tile transposed: Vt[dh][key]
    const bf16* vg = V + (size_t)(b * L + kb + r) * Dm + head * DH + cq;
    v8bf v0 = *(const v8bf*)vg;
    v8bf v1 = *(const v8bf*)(vg + 8);
#pragma unroll
    for (int j = 0; j < 8; ++j) {
      Vt[(cq + j) * LDT + r]     = v0[j];
      Vt[(cq + 8 + j) * LDT + r] = v1[j];
    }
    __syncthreads();

    // S = Q K^T (scale 1/sqrt(64) applied after)
    v8f s0 = {}, s1 = {};
#pragma unroll
    for (int ks = 0; ks < 2; ++ks) {
      v16bf a   = load_a_frag(Qs, LDT, m0, ks * 32, lane);
      v16bf bf0 = load_b_frag(Ks, LDT, n0,      ks * 32, lane);
      v16bf bf1 = load_b_frag(Ks, LDT, n0 + 16, ks * 32, lane);
      s0 = wmma_bf16(a, bf0, s0);
      s1 = wmma_bf16(a, bf1, s1);
    }
#pragma unroll
    for (int rr = 0; rr < 8; ++rr) {
      int lr = m0 + rr + h * 8;
      Ss[lr * 65 + n0 + n]      = s0[rr] * 0.125f;
      Ss[lr * 65 + n0 + 16 + n] = s1[rr] * 0.125f;
    }
    __syncthreads();

    // Online softmax (4 threads per row)
    int srow = tid >> 2, sub = tid & 3;
    float lmax = -3.0e38f;
#pragma unroll
    for (int c = 0; c < 16; ++c)
      lmax = fmaxf(lmax, Ss[srow * 65 + sub * 16 + c]);
    redm[srow * 4 + sub] = lmax;
    __syncthreads();
    if (sub == 0) {
      float rm = fmaxf(fmaxf(redm[srow * 4], redm[srow * 4 + 1]),
                       fmaxf(redm[srow * 4 + 2], redm[srow * 4 + 3]));
      float nm = fmaxf(mrow[srow], rm);
      alphaS[srow] = expf(mrow[srow] - nm);
      mrow[srow] = nm;
    }
    __syncthreads();
    float nm = mrow[srow];
    float ls = 0.f;
#pragma unroll
    for (int c = 0; c < 16; ++c) {
      int cc = sub * 16 + c;
      float p = expf(Ss[srow * 65 + cc] - nm);
      Ps[srow * 66 + cc] = (bf16)p;
      ls += p;
    }
    reds[srow * 4 + sub] = ls;
    __syncthreads();
    if (sub == 0) {
      lrow[srow] = lrow[srow] * alphaS[srow] +
                   (reds[srow * 4] + reds[srow * 4 + 1] +
                    reds[srow * 4 + 2] + reds[srow * 4 + 3]);
    }
    __syncthreads();

    // Rescale O and accumulate O += P V
#pragma unroll
    for (int rr = 0; rr < 8; ++rr) {
      int lr = m0 + rr + h * 8;
      float al = alphaS[lr];
      o0[rr] *= al;
      o1[rr] *= al;
    }
#pragma unroll
    for (int ks = 0; ks < 2; ++ks) {
      v16bf a   = load_a_frag(Ps, 66, m0, ks * 32, lane);
      v16bf bf0 = load_b_frag(Vt, LDT, n0,      ks * 32, lane);
      v16bf bf1 = load_b_frag(Vt, LDT, n0 + 16, ks * 32, lane);
      o0 = wmma_bf16(a, bf0, o0);
      o1 = wmma_bf16(a, bf1, o1);
    }
  }
  __syncthreads();

#pragma unroll
  for (int rr = 0; rr < 8; ++rr) {
    int lr = m0 + rr + h * 8;
    float linv = 1.0f / lrow[lr];
    size_t rowBase = (size_t)(b * L + qBase + lr) * Dm + head * DH;
    ctx[rowBase + n0 + n]      = (bf16)(o0[rr] * linv);
    ctx[rowBase + n0 + 16 + n] = (bf16)(o1[rr] * linv);
  }
}

// ---------------------------------------------------------------------------
// LayerNorm over D=1024, one block per row; writes f32 and optional bf16 copy
// ---------------------------------------------------------------------------
__global__ __launch_bounds__(256) void ln_kernel(
    const float* __restrict__ X, const float* __restrict__ g,
    const float* __restrict__ b, float* __restrict__ outf,
    bf16* __restrict__ outb) {
  int row = blockIdx.x, tid = threadIdx.x;
  const float* xr = X + (size_t)row * 1024;
  float vals[4], s = 0.f, q = 0.f;
#pragma unroll
  for (int i = 0; i < 4; ++i) {
    float v = xr[tid + i * 256];
    vals[i] = v;
    s += v;
    q += v * v;
  }
  __shared__ float rs[256], rq[256];
  rs[tid] = s; rq[tid] = q;
  __syncthreads();
  for (int st = 128; st > 0; st >>= 1) {
    if (tid < st) { rs[tid] += rs[tid + st]; rq[tid] += rq[tid + st]; }
    __syncthreads();
  }
  __shared__ float mu, inv;
  if (tid == 0) {
    mu = rs[0] * (1.0f / 1024.0f);
    float var = rq[0] * (1.0f / 1024.0f) - mu * mu;
    inv = rsqrtf(var + 1e-5f);
  }
  __syncthreads();
#pragma unroll
  for (int i = 0; i < 4; ++i) {
    int c = tid + i * 256;
    float v = (vals[i] - mu) * inv * g[c] + b[c];
    outf[(size_t)row * 1024 + c] = v;
    if (outb) outb[(size_t)row * 1024 + c] = (bf16)v;
  }
}

// ---------------------------------------------------------------------------
// Gate: logits = X @ gate_w + gate_b; top-2 softmax * SCALING scattered to
// dense G[token][8]. One wave per token (wave32 shuffles).
// ---------------------------------------------------------------------------
__global__ __launch_bounds__(256) void gate_topk(
    const float* __restrict__ X, const float* __restrict__ GW,
    const float* __restrict__ GB, float* __restrict__ G) {
  int w = threadIdx.x >> 5, lane = threadIdx.x & 31;
  int token = blockIdx.x * 8 + w;
  float acc[8];
#pragma unroll
  for (int e = 0; e < 8; ++e) acc[e] = 0.f;
  for (int d = lane; d < 1024; d += 32) {
    float xv = X[(size_t)token * 1024 + d];
    const float* gw = GW + (size_t)d * 8;
#pragma unroll
    for (int e = 0; e < 8; ++e) acc[e] += xv * gw[e];
  }
#pragma unroll
  for (int e = 0; e < 8; ++e)
    for (int off = 16; off > 0; off >>= 1)
      acc[e] += __shfl_xor(acc[e], off);
  if (lane == 0) {
    float lg[8];
#pragma unroll
    for (int e = 0; e < 8; ++e) lg[e] = acc[e] + GB[e];
    int i1 = 0;
    for (int e = 1; e < 8; ++e)
      if (lg[e] > lg[i1]) i1 = e;
    int i2 = (i1 == 0) ? 1 : 0;
    for (int e = 0; e < 8; ++e)
      if (e != i1 && lg[e] > lg[i2]) i2 = e;
    float e2 = expf(lg[i2] - lg[i1]);
    float sden = 1.0f + e2;
    float* gr = G + (size_t)token * 8;
#pragma unroll
    for (int e = 0; e < 8; ++e) gr[e] = 0.f;
    gr[i1] = (1.0f / sden) * 2.0f;  // SCALING folded in
    gr[i2] = (e2 / sden) * 2.0f;
  }
}

// ---------------------------------------------------------------------------
// Fused MoE combine: out = layer_out + sum_e G[:,e] * (h_e @ W2_e + b2_e)
// h: [4096, 512] bf16 (col = e*64 + r); W2: [8][64][1024] bf16; G folded
// with SCALING and zeros for unselected experts. WMMA accumulates all 8
// experts into the same f32 fragments (A-tiles pre-scaled by G while staging).
// ---------------------------------------------------------------------------
__global__ __launch_bounds__(256) void moe_combine(
    const bf16* __restrict__ H, const bf16* __restrict__ W2,
    const float* __restrict__ B2, const float* __restrict__ G,
    const float* __restrict__ base, float* __restrict__ out) {
  __shared__ bf16 As[64 * LDT];
  __shared__ bf16 Bt[64 * LDT];
  __shared__ float Gs[64 * 8];
  __shared__ float b2s[8 * 64];

  int tid = threadIdx.x;
  int lane = tid & 31, w = tid >> 5;
  int m0 = (w >> 1) * 16, n0 = (w & 1) * 32;
  int mBase = blockIdx.y * 64, nBase = blockIdx.x * 64;

  for (int i = tid; i < 512; i += 256) {
    Gs[i]  = G[(size_t)(mBase + (i >> 3)) * 8 + (i & 7)];
    b2s[i] = B2[(size_t)(i >> 6) * 1024 + nBase + (i & 63)];
  }
  __syncthreads();

  v8f o0 = {}, o1 = {};
  int r = tid >> 2, cq = (tid & 3) * 16;

  for (int e = 0; e < 8; ++e) {
    // Stage A tile = h[:, e*64 : e*64+64] scaled by G[row, e]
    float gsc = Gs[r * 8 + e];
    const bf16* hg = H + (size_t)(mBase + r) * 512 + e * 64 + cq;
    v8bf a0 = *(const v8bf*)hg;
    v8bf a1 = *(const v8bf*)(hg + 8);
#pragma unroll
    for (int j = 0; j < 8; ++j) {
      As[r * LDT + cq + j]     = (bf16)((float)a0[j] * gsc);
      As[r * LDT + cq + 8 + j] = (bf16)((float)a1[j] * gsc);
    }
    // Stage B tile transposed from W2[e]
    const bf16* wg = W2 + ((size_t)e * 64 + r) * 1024 + nBase + cq;
    v8bf b0 = *(const v8bf*)wg;
    v8bf b1 = *(const v8bf*)(wg + 8);
#pragma unroll
    for (int j = 0; j < 8; ++j) {
      Bt[(cq + j) * LDT + r]     = b0[j];
      Bt[(cq + 8 + j) * LDT + r] = b1[j];
    }
    __syncthreads();
#pragma unroll
    for (int ks = 0; ks < 2; ++ks) {
      v16bf a   = load_a_frag(As, LDT, m0, ks * 32, lane);
      v16bf bf0 = load_b_frag(Bt, LDT, n0,      ks * 32, lane);
      v16bf bf1 = load_b_frag(Bt, LDT, n0 + 16, ks * 32, lane);
      o0 = wmma_bf16(a, bf0, o0);
      o1 = wmma_bf16(a, bf1, o1);
    }
    __syncthreads();
  }

  int n = lane & 15, h = lane >> 4;
#pragma unroll
  for (int rr = 0; rr < 8; ++rr) {
    int lr = m0 + rr + h * 8;
    int gm = mBase + lr;
    int c0l = n0 + n, c1l = n0 + 16 + n;
    float bs0 = 0.f, bs1 = 0.f;
#pragma unroll
    for (int e = 0; e < 8; ++e) {
      float ge = Gs[lr * 8 + e];
      bs0 += ge * b2s[e * 64 + c0l];
      bs1 += ge * b2s[e * 64 + c1l];
    }
    size_t i0 = (size_t)gm * 1024 + nBase + c0l;
    size_t i1 = (size_t)gm * 1024 + nBase + c1l;
    out[i0] = base[i0] + o0[rr] + bs0;
    out[i1] = base[i1] + o1[rr] + bs1;
  }
}

// ---------------------------------------------------------------------------
// Host-side orchestration
// ---------------------------------------------------------------------------
extern "C" void kernel_launch(void* const* d_in, const int* in_sizes, int n_in,
                              void* d_out, int out_size, void* d_ws, size_t ws_size,
                              hipStream_t stream) {
  (void)in_sizes; (void)n_in; (void)out_size; (void)ws_size;
  const int Btok = 4096;     // B*L
  const int D = 1024, FF = 4096, ER = 512;

  const float* x      = (const float*)d_in[0];
  const float* wq     = (const float*)d_in[1];
  const float* wk     = (const float*)d_in[2];
  const float* wv     = (const float*)d_in[3];
  const float* wo     = (const float*)d_in[4];
  const float* bq     = (const float*)d_in[5];
  const float* bk     = (const float*)d_in[6];
  const float* bv     = (const float*)d_in[7];
  const float* bo     = (const float*)d_in[8];
  const float* ln1_g  = (const float*)d_in[9];
  const float* ln1_b  = (const float*)d_in[10];
  const float* w1     = (const float*)d_in[11];
  const float* b1     = (const float*)d_in[12];
  const float* w2     = (const float*)d_in[13];
  const float* b2     = (const float*)d_in[14];
  const float* ln2_g  = (const float*)d_in[15];
  const float* ln2_b  = (const float*)d_in[16];
  const float* adp_w1 = (const float*)d_in[17];
  const float* adp_b1 = (const float*)d_in[18];
  const float* adp_w2 = (const float*)d_in[19];
  const float* adp_b2 = (const float*)d_in[20];
  const float* gate_w = (const float*)d_in[21];
  const float* gate_b = (const float*)d_in[22];
  float* out = (float*)d_out;

  // Workspace arena
  char* ws = (char*)d_ws;
  size_t off = 0;
  auto alloc = [&](size_t bytes) {
    size_t o = off;
    off = (off + bytes + 255) & ~(size_t)255;
    return o;
  };
  bf16* xb   = (bf16*)(ws + alloc((size_t)Btok * D * 2));
  bf16* wqb  = (bf16*)(ws + alloc((size_t)D * D * 2));
  bf16* wkb  = (bf16*)(ws + alloc((size_t)D * D * 2));
  bf16* wvb  = (bf16*)(ws + alloc((size_t)D * D * 2));
  bf16* wob  = (bf16*)(ws + alloc((size_t)D * D * 2));
  bf16* w1b  = (bf16*)(ws + alloc((size_t)D * FF * 2));
  bf16* w2b  = (bf16*)(ws + alloc((size_t)FF * D * 2));
  bf16* aw1b = (bf16*)(ws + alloc((size_t)D * ER * 2));
  bf16* aw2b = (bf16*)(ws + alloc((size_t)ER * D * 2));
  // Q/K/V/ctx region, later reused for FFN intermediate (same 32MB footprint)
  bf16* qkv  = (bf16*)(ws + alloc((size_t)Btok * FF * 2));
  bf16* qb   = qkv;
  bf16* kb   = qkv + (size_t)Btok * D;
  bf16* vb   = qkv + (size_t)2 * Btok * D;
  bf16* ctxb = qkv + (size_t)3 * Btok * D;
  bf16* interb = qkv;  // reuse after attention+O-proj complete
  float* pre    = (float*)(ws + alloc((size_t)Btok * D * 4));  // shared attn/ffn pre-LN
  float* attn_f = (float*)(ws + alloc((size_t)Btok * D * 4));
  bf16*  attn_b = (bf16*)(ws + alloc((size_t)Btok * D * 2));
  float* layer_f = (float*)(ws + alloc((size_t)Btok * D * 4));
  bf16*  hb     = (bf16*)(ws + alloc((size_t)Btok * ER * 2));
  float* G      = (float*)(ws + alloc((size_t)Btok * 8 * 4));

  // 1) fp32 -> bf16 conversions
  auto cvt = [&](const float* s, bf16* d, int n) {
    int blocks = (n + 1023) / 1024;
    if (blocks > 2048) blocks = 2048;
    cvt_f32_bf16<<<blocks, 256, 0, stream>>>(s, d, n);
  };
  cvt(x, xb, Btok * D);
  cvt(wq, wqb, D * D);
  cvt(wk, wkb, D * D);
  cvt(wv, wvb, D * D);
  cvt(wo, wob, D * D);
  cvt(w1, w1b, D * FF);
  cvt(w2, w2b, FF * D);
  cvt(adp_w1, aw1b, D * ER);
  cvt(adp_w2, aw2b, ER * D);

  // 2) QKV projections (bf16 outputs)  [block tile 128x64]
  dim3 gD(D / 64, Btok / 128);
  gemm_bf16<<<gD, 256, 0, stream>>>(xb, wqb, bq, nullptr, nullptr, qb, Btok, D, D, 0);
  gemm_bf16<<<gD, 256, 0, stream>>>(xb, wkb, bk, nullptr, nullptr, kb, Btok, D, D, 0);
  gemm_bf16<<<gD, 256, 0, stream>>>(xb, wvb, bv, nullptr, nullptr, vb, Btok, D, D, 0);

  // 3) Fused attention -> ctx (bf16)
  attn_kernel<<<dim3(16, 64), 256, 0, stream>>>(qb, kb, vb, ctxb);

  // 4) O-projection + residual(x) -> pre; LN1 -> attn_f / attn_b
  gemm_bf16<<<gD, 256, 0, stream>>>(ctxb, wob, bo, x, pre, nullptr, Btok, D, D, 0);
  ln_kernel<<<Btok, 256, 0, stream>>>(pre, ln1_g, ln1_b, attn_f, attn_b);

  // 5) FFN: gelu(attn @ w1 + b1) -> interb (bf16); @ w2 + b2 + attn_f -> pre
  gemm_bf16<<<dim3(FF / 64, Btok / 128), 256, 0, stream>>>(
      attn_b, w1b, b1, nullptr, nullptr, interb, Btok, FF, D, 1);
  gemm_bf16<<<gD, 256, 0, stream>>>(interb, w2b, b2, attn_f, pre, nullptr, Btok, D, FF, 0);
  ln_kernel<<<Btok, 256, 0, stream>>>(pre, ln2_g, ln2_b, layer_f, nullptr);

  // 6) Adapter: h = gelu(attn @ adp_w1 + adp_b1) (bf16)
  gemm_bf16<<<dim3(ER / 64, Btok / 128), 256, 0, stream>>>(
      attn_b, aw1b, adp_b1, nullptr, nullptr, hb, Btok, ER, D, 1);

  // 7) Gate top-2 softmax * SCALING -> dense G
  gate_topk<<<Btok / 8, 256, 0, stream>>>(attn_f, gate_w, gate_b, G);

  // 8) Fused MoE combine + final add -> d_out
  moe_combine<<<dim3(16, 64), 256, 0, stream>>>(hb, aw2b, adp_b2, G, layer_f, out);
}
